// NetVLAD_12713103196706
// MI455X (gfx1250) — matrix operand order repaired and checked by reading, not compiled, not run on hardware
//
#include <hip/hip_runtime.h>
#include <hip/hip_bf16.h>

typedef __attribute__((ext_vector_type(16))) _Float16 v16h;
typedef __attribute__((ext_vector_type(8)))  float    v8f;

namespace {
constexpr int NIMG   = 64;
constexpr int CDIM   = 128;
constexpr int PDIM   = 4096;
constexpr int KCL    = 64;
constexpr int PRANGE = 1024;          // pixels per workgroup
constexpr int PTILE  = 64;            // pixels per subtile
constexpr int NSUB   = PRANGE / PTILE;// 16
constexpr int WSTR   = 132;           // conv_w LDS row stride (halves): 66 dwords -> conflict-free A gathers
constexpr int B1STR  = 66;            // xf [c][p] stride (halves): 33 dwords, gcd(33,64)=1
constexpr int T2STR  = 132;           // xf [p][c] stride (halves)
constexpr int LGSTR  = 66;            // logits stride (floats) -- aliases the [p][c] tile
constexpr int ASTR2  = 66;            // a[k][p] stride (halves)
constexpr float EPSF = 1e-12f;
}

// ---------------------------------------------------------------------------
// Kernel 1: normalize + GEMM1 (logits) + softmax + GEMM2 (vlad partials)
// grid = (4 p-ranges, 64 images), block = 512 threads (16 wave32)
// ---------------------------------------------------------------------------
__global__ __launch_bounds__(512)
void netvlad_main(const float* __restrict__ x,       // [N,C,P]
                  const float* __restrict__ conv_w,  // [K,C]
                  const float* __restrict__ conv_b,  // [K]
                  float* __restrict__ vlad_ws,       // [N,C,K] f32 partial sums (pre-zeroed)
                  float* __restrict__ asum_ws)       // [N,K]   f32 partial sums (pre-zeroed)
{
  __shared__ _Float16 wF16[KCL * WSTR];       // conv_w as f16, [k][c]
  __shared__ _Float16 tileB1[CDIM * B1STR];   // xf tile, [c][p]   (GEMM2 A source)
  __shared__ __align__(16) char uBraw[KCL * LGSTR * 4]; // xf [p][c] tile, later aliased as logits
  __shared__ _Float16 aKP[KCL * ASTR2];       // softmax weights, [k][p]
  __shared__ float convB[KCL];
  __shared__ float sred[512];
  __shared__ float colstat[PTILE];

  _Float16* t2 = (_Float16*)uBraw;            // [p][c], stride T2STR
  float*    lg = (float*)uBraw;               // [k][p], stride LGSTR (after GEMM1 reads done)

  const int tid  = threadIdx.x;
  const int lane = tid & 31;
  const int wave = tid >> 5;                  // 0..15
  const int n    = blockIdx.y;
  const int p0   = blockIdx.x * PRANGE;
  const float* xn = x + (size_t)n * CDIM * PDIM;

  // WMMA lane->element mappings (ISA 7.12.2, wave32)
  const int fm  = lane & 15;                  // A-matrix row
  const int fkb = (lane >> 4) << 3;           // A K-base (0 / 8)
  const int fn  = lane & 15;                  // B / D column
  const int fbk = (lane >> 4) << 4;           // unused (0 / 16); B gathers use fkb like A
  const int fdm = (lane >> 4) << 3;           // D row base (0 / 8)

  // stage conv_w (f16) and conv_b into LDS
  for (int e = tid; e < KCL * CDIM; e += 512) {
    const int k = e >> 7, c = e & 127;
    wF16[k * WSTR + c] = (_Float16)conv_w[e];
  }
  if (tid < KCL) convB[tid] = conv_b[tid];

  // GEMM1 tile for this wave: k-tile x p-tile of [64 x 64] logits
  const int kt1 = wave >> 2;                  // 0..3
  const int pt1 = wave & 3;                   // 0..3
  // GEMM2 (transposed) tiles for this wave: vladT[c,k]; (ctm, ctm+4) x ktn
  const int ctm = wave >> 2;                  // 0..3  (+4 for second tile)
  const int ktn = wave & 3;                   // 0..3

  v8f accV0, accV1;
  #pragma unroll
  for (int i = 0; i < 8; ++i) { accV0[i] = 0.f; accV1[i] = 0.f; }
  float asum_acc = 0.f;

  for (int ps = 0; ps < NSUB; ++ps) {
    const int pb = ps * PTILE;
    const int pl = tid & 63;                  // this thread's pixel within the subtile

    // ---- prefetch next subtile: 128 threads, one cacheline per channel row ----
    if (ps + 1 < NSUB && tid < CDIM)
      __builtin_prefetch(xn + (size_t)tid * PDIM + p0 + pb + PTILE, 0, 1);

    // ---- per-pixel channel sum-of-squares (x kept in registers) ----
    float xv[16];
    float ss = 0.f;
    #pragma unroll
    for (int i = 0; i < 16; ++i) {
      const int c = (tid >> 6) + i * 8;       // 16 channels per thread
      const float vv = xn[(size_t)c * PDIM + p0 + pb + pl];
      xv[i] = vv; ss += vv * vv;
    }
    sred[tid] = ss;                           // 8 partials per pixel
    __syncthreads();
    if (tid < PTILE) {
      float s = 0.f;
      #pragma unroll
      for (int j = 0; j < 8; ++j) s += sred[tid + 64 * j];
      colstat[tid] = 1.f / fmaxf(sqrtf(s), EPSF);
    }
    __syncthreads();

    // ---- write normalized f16 tiles in both orientations ----
    const float inp = colstat[pl];
    #pragma unroll
    for (int i = 0; i < 16; ++i) {
      const int c = (tid >> 6) + i * 8;
      const _Float16 hv = (_Float16)(xv[i] * inp);
      tileB1[c * B1STR + pl] = hv;            // [c][p]
      t2[pl * T2STR + c]     = hv;            // [p][c]
    }
    __syncthreads();

    // ---- GEMM1: logits[k,p] = W[k,:] . xf[:,p]  (M=k, N=p, inner=c) ----
    v8f acc1;
    #pragma unroll
    for (int i = 0; i < 8; ++i) acc1[i] = 0.f;
    #pragma unroll
    for (int cc = 0; cc < 4; ++cc) {
      v16h A, B;
      const _Float16* ar = &wF16[(kt1 * 16 + fm) * WSTR + cc * 32];
      #pragma unroll
      for (int v = 0; v < 8; ++v) {
        const int k0 = ((v & 4) << 2) + fkb + ((v & 3) << 1);
        A[2 * v]     = ar[k0];
        A[2 * v + 1] = ar[k0 + 1];
      }
      const _Float16* br = &t2[(pt1 * 16 + fn) * T2STR + cc * 32 + fkb];
      #pragma unroll
      for (int h = 0; h < 16; ++h) B[h] = br[h + ((h >> 3) << 3)];
      acc1 = __builtin_amdgcn_wmma_f32_16x16x32_f16(false, A, false, B,
                                                    (short)0, acc1, false, false);
    }
    __syncthreads();                          // all t2 reads done -> reuse storage as logits
    #pragma unroll
    for (int v = 0; v < 8; ++v) {
      const int mr = kt1 * 16 + fdm + v;
      lg[mr * LGSTR + pt1 * 16 + fn] = acc1[v] + convB[mr];
    }
    __syncthreads();

    // ---- softmax over k (64) per pixel, 4 threads per column ----
    if (tid < 256) {
      const int p = tid & 63, q = tid >> 6;
      float m = -3.4e38f;
      for (int k = q * 16; k < q * 16 + 16; ++k)
        m = fmaxf(m, lg[k * LGSTR + p]);
      sred[tid] = m;
    }
    __syncthreads();
    if (tid < PTILE)
      colstat[tid] = fmaxf(fmaxf(sred[tid], sred[tid + 64]),
                           fmaxf(sred[tid + 128], sred[tid + 192]));
    __syncthreads();
    if (tid < 256) {
      const int p = tid & 63, q = tid >> 6;
      const float m = colstat[p];
      float s = 0.f;
      for (int k = q * 16; k < q * 16 + 16; ++k) {
        const float e = __expf(lg[k * LGSTR + p] - m);
        aKP[k * ASTR2 + p] = (_Float16)e;
        s += e;
      }
      sred[tid] = s;
    }
    __syncthreads();
    if (tid < PTILE)
      colstat[tid] = 1.f / (sred[tid] + sred[tid + 64] + sred[tid + 128] + sred[tid + 192]);
    __syncthreads();
    if (tid < 256) {
      const int p = tid & 63, q = tid >> 6;
      const float is = colstat[p];
      for (int k = q * 16; k < q * 16 + 16; ++k)
        aKP[k * ASTR2 + p] = (_Float16)((float)aKP[k * ASTR2 + p] * is);
    }
    __syncthreads();

    // ---- per-cluster softmax mass (for centroid term) ----
    {
      const int kk = tid >> 3, pp = (tid & 7) * 8;
      float s = 0.f;
      #pragma unroll
      for (int j = 0; j < 8; ++j) s += (float)aKP[kk * ASTR2 + pp + j];
      asum_acc += s;
    }

    // ---- GEMM2 (transposed): vladT[c,k] += xf[c,p-range] . a^T[p,k] ----
    #pragma unroll
    for (int pc = 0; pc < 2; ++pc) {
      v16h A0, A1, B;
      const _Float16* a0 = &tileB1[(ctm * 16 + fm) * B1STR + pc * 32];
      const _Float16* a1 = &tileB1[((ctm + 4) * 16 + fm) * B1STR + pc * 32];
      #pragma unroll
      for (int v = 0; v < 8; ++v) {
        const int k0 = ((v & 4) << 2) + fkb + ((v & 3) << 1);
        A0[2 * v] = a0[k0]; A0[2 * v + 1] = a0[k0 + 1];
        A1[2 * v] = a1[k0]; A1[2 * v + 1] = a1[k0 + 1];
      }
      const _Float16* br = &aKP[(ktn * 16 + fn) * ASTR2 + pc * 32 + fkb];
      #pragma unroll
      for (int h = 0; h < 16; ++h) B[h] = br[h + ((h >> 3) << 3)];
      accV0 = __builtin_amdgcn_wmma_f32_16x16x32_f16(false, A0, false, B,
                                                     (short)0, accV0, false, false);
      accV1 = __builtin_amdgcn_wmma_f32_16x16x32_f16(false, A1, false, B,
                                                     (short)0, accV1, false, false);
    }
    __syncthreads();
  }

  // ---- flush per-wave accumulators: vlad_ws is [n][c][k] (coalesced atomics) ----
  #pragma unroll
  for (int v = 0; v < 8; ++v) {
    const int c0 = ctm * 16 + fdm + v;
    const int c1 = (ctm + 4) * 16 + fdm + v;
    const int kg = ktn * 16 + fn;
    atomicAdd(&vlad_ws[((size_t)n * CDIM + c0) * KCL + kg], accV0[v]);
    atomicAdd(&vlad_ws[((size_t)n * CDIM + c1) * KCL + kg], accV1[v]);
  }
  atomicAdd(&asum_ws[n * KCL + (tid >> 3)], asum_acc);
}

// ---------------------------------------------------------------------------
// Kernel 2: centroid correction + intra-norm + global norm -> out [N, K*C]
// ---------------------------------------------------------------------------
__global__ __launch_bounds__(512)
void netvlad_finalize(const float* __restrict__ vlad_ws,   // [N,C,K]
                      const float* __restrict__ asum_ws,   // [N,K]
                      const float* __restrict__ centroids, // [K,C]
                      float* __restrict__ out)             // [N, K*C]
{
  __shared__ float vl[KCL * CDIM];
  __shared__ float asumS[KCL];
  __shared__ float rinv[KCL];
  __shared__ float gs, gsinv;

  const int tid = threadIdx.x;
  const int n   = blockIdx.x;

  if (tid < KCL) asumS[tid] = asum_ws[n * KCL + tid];
  if (tid == 0)  gs = 0.f;
  __syncthreads();

  #pragma unroll
  for (int i = 0; i < 16; ++i) {
    const int idx = tid + i * 512;            // k*128 + c
    const int k = idx >> 7, c = idx & 127;
    vl[idx] = vlad_ws[((size_t)n * CDIM + c) * KCL + k]
            - asumS[k] * centroids[k * CDIM + c];
  }
  __syncthreads();

  if (tid < KCL) {
    float s = 0.f;
    for (int c = 0; c < CDIM; ++c) { const float v = vl[tid * CDIM + c]; s += v * v; }
    rinv[tid] = 1.f / fmaxf(sqrtf(s), EPSF);
  }
  __syncthreads();

  float part = 0.f;
  #pragma unroll
  for (int i = 0; i < 16; ++i) {
    const int idx = tid + i * 512;
    const float v = vl[idx] * rinv[idx >> 7];
    vl[idx] = v;
    part += v * v;
  }
  atomicAdd(&gs, part);
  __syncthreads();
  if (tid == 0) gsinv = 1.f / fmaxf(sqrtf(gs), EPSF);
  __syncthreads();

  #pragma unroll
  for (int i = 0; i < 16; ++i) {
    const int idx = tid + i * 512;
    out[(size_t)n * (KCL * CDIM) + idx] = vl[idx] * gsinv;
  }
}

// ---------------------------------------------------------------------------
extern "C" void kernel_launch(void* const* d_in, const int* in_sizes, int n_in,
                              void* d_out, int out_size, void* d_ws, size_t ws_size,
                              hipStream_t stream) {
  const float* x         = (const float*)d_in[0];   // [64,128,64,64]
  // d_in[1] = xyz (unused by the reference)
  const float* centroids = (const float*)d_in[2];   // [64,128]
  const float* conv_w    = (const float*)d_in[3];   // [64,128]
  const float* conv_b    = (const float*)d_in[4];   // [64]
  float* out = (float*)d_out;

  float* vlad_ws = (float*)d_ws;                               // [N,C,K] = 2 MB
  float* asum_ws = vlad_ws + (size_t)NIMG * CDIM * KCL;        // [N,K]   = 16 KB
  const size_t zero_bytes =
      ((size_t)NIMG * CDIM * KCL + (size_t)NIMG * KCL) * sizeof(float);
  (void)in_sizes; (void)n_in; (void)out_size; (void)ws_size;

  hipMemsetAsync(d_ws, 0, zero_bytes, stream);     // capture-safe memset node
  netvlad_main<<<dim3(PDIM / PRANGE, NIMG), 512, 0, stream>>>(
      x, conv_w, conv_b, vlad_ws, asum_ws);
  netvlad_finalize<<<NIMG, 512, 0, stream>>>(vlad_ws, asum_ws, centroids, out);
}
